// Gemma4Model_54803782697632
// MI455X (gfx1250) — matrix operand order repaired and checked by reading, not compiled, not run on hardware
//
#include <hip/hip_runtime.h>

typedef __attribute__((ext_vector_type(16))) __bf16 v16bf;
typedef __attribute__((ext_vector_type(8)))  float  v8f;
typedef __attribute__((ext_vector_type(4)))  int    v4i;

#define SEQ    4096
#define DMODEL 2048
#define NH     8
#define NKV    4
#define DHEAD  256
#define WIN    1024
#define EPSILON 1e-6f

#if __has_builtin(__builtin_amdgcn_global_load_async_to_lds_b128) && \
    __has_builtin(__builtin_amdgcn_s_wait_asynccnt)
#define HAVE_ASYNC 1
#else
#define HAVE_ASYNC 0
#endif

typedef __attribute__((address_space(3))) void* as3_void;
typedef __attribute__((address_space(3))) v4i*  as3_v4i;

// 16-byte global -> LDS copy via gfx1250 GLOBAL_LOAD_ASYNC_TO_LDS_B128 (ASYNCcnt).
__device__ __forceinline__ void copy16(const __bf16* g, __bf16* l) {
#if HAVE_ASYNC
  __builtin_amdgcn_global_load_async_to_lds_b128((v4i*)g, (as3_v4i)(as3_void)l,
                                                 0, 0);
#else
  *(uint4*)(void*)l = *(const uint4*)(const void*)g;
#endif
}
__device__ __forceinline__ void copy_wait() {
#if HAVE_ASYNC
  __builtin_amdgcn_s_wait_asynccnt(0);
#endif
}

// A operand as this code loads it (contiguous k, not ISA table order): lane=16*kh+m, elem e -> K = 16*kh+e
// => one contiguous 32B run (matches B).
// B operand as loaded here: lane=16*kh+n, elem e -> K = 16*kh+e => one contiguous 32B run.
union Frag16 { uint4 u[2]; v16bf v; };

__device__ __forceinline__ v8f wmma_bf16(v16bf a, v16bf b, v8f c) {
  return __builtin_amdgcn_wmma_f32_16x16x32_bf16(false, a, false, b, (short)0, c,
                                                 false, false);
}

// ---------------------------------------------------------------------------
// Kernel 0a: plain f32 -> bf16 conversion (hidden states).
// ---------------------------------------------------------------------------
__global__ __launch_bounds__(256) void k_cvt(const float* __restrict__ src,
                                             __bf16* __restrict__ dst, int n) {
  int i = (blockIdx.x * 256 + threadIdx.x) * 8;
  if (i + 8 <= n) {
    float4 f0 = *(const float4*)&src[i];
    float4 f1 = *(const float4*)&src[i + 4];
    union { __bf16 h[8]; uint4 u; } t;
    t.h[0] = (__bf16)f0.x; t.h[1] = (__bf16)f0.y;
    t.h[2] = (__bf16)f0.z; t.h[3] = (__bf16)f0.w;
    t.h[4] = (__bf16)f1.x; t.h[5] = (__bf16)f1.y;
    t.h[6] = (__bf16)f1.z; t.h[7] = (__bf16)f1.w;
    *(uint4*)(void*)&dst[i] = t.u;
  }
}

// ---------------------------------------------------------------------------
// Kernel 0b: f32 -> bf16 convert + transpose: dst[n][k] = (bf16)src[k][n].
// grid = (N/256, K/8); lanes run over n => coalesced reads; b128 writes.
// ---------------------------------------------------------------------------
__global__ __launch_bounds__(256) void k_cvt_t(const float* __restrict__ src,
                                               __bf16* __restrict__ dst,
                                               int N, int K) {
  int n = blockIdx.x * 256 + threadIdx.x;
  int k0 = blockIdx.y * 8;
  union { __bf16 h[8]; uint4 u; } t;
#pragma unroll
  for (int j = 0; j < 8; ++j)
    t.h[j] = (__bf16)src[(size_t)(k0 + j) * N + n];
  *(uint4*)(void*)&dst[(size_t)n * K + k0] = t.u;
}

// ---------------------------------------------------------------------------
// Kernel 1: QKV projection (K-major bf16 weights, bf16 activations)
//           + RMSNorm + RoPE.
// grid = (SEQ/64, 16), block = 256.  64 seq rows x 256 cols (one head).
// Wave: 32-wide n-slice x 4 m-tiles => 8 accumulators, 16 WMMA / 64-K stage.
// V head output is written transposed: vt[d][s].
// ---------------------------------------------------------------------------
__global__ __launch_bounds__(256) void k_qkv(
    const __bf16* __restrict__ hsb, const float* __restrict__ cosb,
    const float* __restrict__ sinb,
    const __bf16* __restrict__ WqT, const __bf16* __restrict__ WkT,
    const __bf16* __restrict__ WvT,
    const float* __restrict__ qnw, const float* __restrict__ knw,
    __bf16* __restrict__ qb, __bf16* __restrict__ kb, __bf16* __restrict__ vt)
{
  const int tid = threadIdx.x;
  const int wave = tid >> 5, lane = tid & 31;
  const int m16 = lane & 15, kh = lane >> 4;
  const int row0 = blockIdx.x * 64;
  const int slot = blockIdx.y;

  const __bf16* WT; const float* nw; __bf16* outp;
  int col0, ostride; bool rope;
  if (slot < NH) {
    WT = WqT; col0 = slot * DHEAD;
    outp = qb; ostride = NH * DHEAD; nw = qnw; rope = true;
  } else if (slot < NH + NKV) {
    WT = WkT; col0 = (slot - NH) * DHEAD;
    outp = kb; ostride = NKV * DHEAD; nw = knw; rope = true;
  } else {
    WT = WvT; col0 = (slot - NH - NKV) * DHEAD;
    outp = vt; ostride = 0; nw = nullptr; rope = false;
  }

  __shared__ __attribute__((aligned(16))) char smem[8 * 1024 + 32 * 1024];
  __bf16* As  = (__bf16*)smem;              // 64 x 64
  __bf16* BsT = (__bf16*)(smem + 8 * 1024); // 256n x 64k
  __shared__ float rowss[64];

  const int n0 = wave * 32;
  v8f acc[4][2];
#pragma unroll
  for (int mt = 0; mt < 4; ++mt) { acc[mt][0] = (v8f){}; acc[mt][1] = (v8f){}; }

  for (int k0 = 0; k0 < DMODEL; k0 += 64) {
#pragma unroll
    for (int j = 0; j < 2; ++j) {  // stage A: 64x64 bf16 (async b128)
      int idx = tid + j * 256;
      int r = idx >> 3, c = (idx & 7) * 8;
      copy16(&hsb[(size_t)(row0 + r) * DMODEL + k0 + c], &As[r * 64 + c]);
    }
#pragma unroll
    for (int j = 0; j < 8; ++j) {  // stage B^T: 256n x 64k (async b128)
      int idx = tid + j * 256;
      int n = idx >> 3, c = (idx & 7) * 8;
      copy16(&WT[(size_t)(col0 + n) * DMODEL + k0 + c], &BsT[n * 64 + c]);
    }
    copy_wait();
    __syncthreads();

#pragma unroll
    for (int kk = 0; kk < 64; kk += 32) {
      Frag16 b0, b1;
      b0.u[0] = *(const uint4*)(const void*)&BsT[(n0 + m16) * 64 + kk + 16 * kh];
      b0.u[1] = *(const uint4*)(const void*)&BsT[(n0 + m16) * 64 + kk + 16 * kh + 8];
      b1.u[0] = *(const uint4*)(const void*)&BsT[(n0 + 16 + m16) * 64 + kk + 16 * kh];
      b1.u[1] = *(const uint4*)(const void*)&BsT[(n0 + 16 + m16) * 64 + kk + 16 * kh + 8];
#pragma unroll
      for (int mt = 0; mt < 4; ++mt) {  // B reused across 4 m-tiles
        Frag16 a;
        a.u[0] = *(const uint4*)(const void*)&As[(mt * 16 + m16) * 64 + kk + 16 * kh];
        a.u[1] = *(const uint4*)(const void*)&As[(mt * 16 + m16) * 64 + kk + 16 * kh + 8];
        acc[mt][0] = wmma_bf16(a.v, b0.v, acc[mt][0]);
        acc[mt][1] = wmma_bf16(a.v, b1.v, acc[mt][1]);
      }
    }
    __syncthreads();
  }

  // RMSNorm row sums via ds_add_f32
  if (tid < 64) rowss[tid] = 0.f;
  __syncthreads();
#pragma unroll
  for (int mt = 0; mt < 4; ++mt)
#pragma unroll
    for (int r = 0; r < 8; ++r)
      atomicAdd(&rowss[mt * 16 + r + 8 * kh],
                acc[mt][0][r] * acc[mt][0][r] + acc[mt][1][r] * acc[mt][1][r]);
  __syncthreads();

  __bf16* ybuf = BsT;  // alias: 64x256 bf16 = 32KB (GEMM staging done)
#pragma unroll
  for (int mt = 0; mt < 4; ++mt)
#pragma unroll
    for (int r = 0; r < 8; ++r) {
      int m = mt * 16 + r + 8 * kh;
      float sc = rsqrtf(rowss[m] * (1.0f / DHEAD) + EPSILON);
      float w0 = nw ? nw[n0 + m16] : 1.0f;
      float w1 = nw ? nw[n0 + 16 + m16] : 1.0f;
      ybuf[m * 256 + n0 + m16]      = (__bf16)(acc[mt][0][r] * sc * w0);
      ybuf[m * 256 + n0 + 16 + m16] = (__bf16)(acc[mt][1][r] * sc * w1);
    }
  __syncthreads();

  for (int idx = tid; idx < 64 * 256; idx += 256) {
    int m = idx >> 8, n = idx & 255;
    int s = row0 + m;
    float y = (float)ybuf[idx];
    if (rope) {
      float cv = cosb[(size_t)s * DHEAD + n];
      float sv = sinb[(size_t)s * DHEAD + n];
      float pr = (float)ybuf[m * 256 + (n ^ 128)];
      float partner = (n < 128) ? -pr : pr;
      outp[(size_t)s * ostride + col0 + n] = (__bf16)(y * cv + partner * sv);
    } else {
      outp[(size_t)(col0 + n) * SEQ + s] = (__bf16)y;  // V transposed
    }
  }
}

// ---------------------------------------------------------------------------
// Kernel 2: sliding-window causal GQA flash attention, 32 queries / WG.
// grid = (SEQ/32, NH), block = 256 (8 waves).
// ---------------------------------------------------------------------------
__global__ __launch_bounds__(256) void k_attn(
    const __bf16* __restrict__ qbuf, const __bf16* __restrict__ kbuf,
    const __bf16* __restrict__ vt, const int* __restrict__ doc,
    __bf16* __restrict__ ctx)
{
  const int tid = threadIdx.x;
  const int wave = tid >> 5, lane = tid & 31;
  const int m16 = lane & 15, kh = lane >> 4;
  const int q0 = blockIdx.x * 32;
  const int h = blockIdx.y;
  const int kvh = h >> 1;

  __shared__ __attribute__((aligned(16))) __bf16 Qs[32 * 256];
  __shared__ __attribute__((aligned(16))) __bf16 Ks[32 * 256];
  __shared__ __attribute__((aligned(16))) __bf16 VsT[256 * 32];  // [d][key]
  __shared__ float  Ssc[32 * 32];
  __shared__ __attribute__((aligned(16))) __bf16 Ps[32 * 32];
  __shared__ float mrow[32], lrow[32], scl[32];

#pragma unroll
  for (int j = 0; j < 4; ++j) {  // stage Q (async b128)
    int idx = tid + j * 256;
    int r = idx >> 5, c = (idx & 31) * 8;
    copy16(&qbuf[(size_t)(q0 + r) * (NH * DHEAD) + h * DHEAD + c], &Qs[r * 256 + c]);
  }
  if (tid < 32) { mrow[tid] = -__builtin_inff(); lrow[tid] = 0.f; }

  const int d0 = wave * 32;
  const int qm = wave & 1, qn = (wave >> 1) & 1, khf = wave >> 2;
  v8f acc00 = {}, acc01 = {}, acc10 = {}, acc11 = {};

  const int qlast = q0 + 31;
  const int kstart = (qlast >= WIN) ? ((qlast - WIN + 1) & ~31) : 0;

  for (int kb0 = kstart; kb0 <= qlast; kb0 += 32) {
    __syncthreads();  // prior-iteration readers done before we overwrite
#pragma unroll
    for (int j = 0; j < 4; ++j) {  // stage K rows (async b128)
      int idx = tid + j * 256;
      int r = idx >> 5, c = (idx & 31) * 8;
      copy16(&kbuf[(size_t)(kb0 + r) * (NKV * DHEAD) + kvh * DHEAD + c],
             &Ks[r * 256 + c]);
    }
#pragma unroll
    for (int j = 0; j < 4; ++j) {  // stage V^T rows (async b128)
      int idx = tid + j * 256;
      int d = idx >> 2, c = (idx & 3) * 8;
      copy16(&vt[(size_t)(kvh * DHEAD + d) * SEQ + kb0 + c], &VsT[d * 32 + c]);
    }
    for (int i = tid; i < 32 * 32; i += 256) Ssc[i] = 0.f;
    copy_wait();
    __syncthreads();

    // Q.K^T quadrant: rows qm*16.., keys qn*16.., K-dims [khf*128, +128)
    {
      v8f ps = {};
#pragma unroll
      for (int kk = 0; kk < 128; kk += 32) {
        int kd = khf * 128 + kk;
        Frag16 fa, fb;
        fa.u[0] = *(const uint4*)(const void*)&Qs[(qm * 16 + m16) * 256 + kd + 16 * kh];
        fa.u[1] = *(const uint4*)(const void*)&Qs[(qm * 16 + m16) * 256 + kd + 16 * kh + 8];
        fb.u[0] = *(const uint4*)(const void*)&Ks[(qn * 16 + m16) * 256 + kd + 16 * kh];
        fb.u[1] = *(const uint4*)(const void*)&Ks[(qn * 16 + m16) * 256 + kd + 16 * kh + 8];
        ps = wmma_bf16(fa.v, fb.v, ps);
      }
#pragma unroll
      for (int r = 0; r < 8; ++r)
        atomicAdd(&Ssc[(qm * 16 + r + 8 * kh) * 32 + qn * 16 + m16], ps[r]);
    }
    __syncthreads();

    if (tid < 32) {  // online softmax, one thread per query row
      int m = tid, qs = q0 + m, qd = doc[qs];
      float mx = -__builtin_inff();
      for (int j = 0; j < 32; ++j) {
        int ks = kb0 + j;
        bool ok = (ks <= qs) && (qs - ks < WIN) && (doc[ks] == qd);
        float sv = ok ? Ssc[m * 32 + j] : -__builtin_inff();
        Ssc[m * 32 + j] = sv;
        mx = fmaxf(mx, sv);
      }
      float mold = mrow[m];
      float mnew = fmaxf(mold, mx);
      float corr = (mnew > mold) ? __expf(mold - mnew) : 1.0f;
      float ls = 0.f;
      for (int j = 0; j < 32; ++j) {
        float sv = Ssc[m * 32 + j];
        float p = (sv == -__builtin_inff()) ? 0.f : __expf(sv - mnew);
        Ps[m * 32 + j] = (__bf16)p;
        ls += p;
      }
      mrow[m] = mnew;
      lrow[m] = lrow[m] * corr + ls;
      scl[m] = corr;
    }
    __syncthreads();

#pragma unroll
    for (int r = 0; r < 8; ++r) {
      float c0 = scl[r + 8 * kh], c1 = scl[16 + r + 8 * kh];
      acc00[r] *= c0; acc01[r] *= c0;
      acc10[r] *= c1; acc11[r] *= c1;
    }
    Frag16 pa0, pa1, fv0, fv1;
    pa0.u[0] = *(const uint4*)(const void*)&Ps[m16 * 32 + 16 * kh];
    pa0.u[1] = *(const uint4*)(const void*)&Ps[m16 * 32 + 16 * kh + 8];
    pa1.u[0] = *(const uint4*)(const void*)&Ps[(m16 + 16) * 32 + 16 * kh];
    pa1.u[1] = *(const uint4*)(const void*)&Ps[(m16 + 16) * 32 + 16 * kh + 8];
    fv0.u[0] = *(const uint4*)(const void*)&VsT[(d0 + m16) * 32 + 16 * kh];
    fv0.u[1] = *(const uint4*)(const void*)&VsT[(d0 + m16) * 32 + 16 * kh + 8];
    fv1.u[0] = *(const uint4*)(const void*)&VsT[(d0 + 16 + m16) * 32 + 16 * kh];
    fv1.u[1] = *(const uint4*)(const void*)&VsT[(d0 + 16 + m16) * 32 + 16 * kh + 8];
    acc00 = wmma_bf16(pa0.v, fv0.v, acc00);
    acc01 = wmma_bf16(pa0.v, fv1.v, acc01);
    acc10 = wmma_bf16(pa1.v, fv0.v, acc10);
    acc11 = wmma_bf16(pa1.v, fv1.v, acc11);
  }
  __syncthreads();

#pragma unroll
  for (int r = 0; r < 8; ++r) {
    int ma = r + 8 * kh, mb = 16 + r + 8 * kh;
    float ia = 1.0f / lrow[ma], ib = 1.0f / lrow[mb];
    size_t ba = (size_t)(q0 + ma) * (NH * DHEAD) + h * DHEAD;
    size_t bb = (size_t)(q0 + mb) * (NH * DHEAD) + h * DHEAD;
    ctx[ba + d0 + m16]      = (__bf16)(acc00[r] * ia);
    ctx[ba + d0 + 16 + m16] = (__bf16)(acc01[r] * ia);
    ctx[bb + d0 + m16]      = (__bf16)(acc10[r] * ib);
    ctx[bb + d0 + 16 + m16] = (__bf16)(acc11[r] * ib);
  }
}

// ---------------------------------------------------------------------------
// Kernel 3: output projection ctx(bf16) @ Wo^T(bf16, K-major) -> f32.
// grid = (SEQ/64, DMODEL/256), block = 256.  64x256 tile per workgroup.
// ---------------------------------------------------------------------------
__global__ __launch_bounds__(256) void k_oproj(
    const __bf16* __restrict__ ctx, const __bf16* __restrict__ WoT,
    float* __restrict__ out)
{
  const int tid = threadIdx.x;
  const int wave = tid >> 5, lane = tid & 31;
  const int m16 = lane & 15, kh = lane >> 4;
  const int row0 = blockIdx.x * 64;
  const int col0 = blockIdx.y * 256;
  const int n0 = wave * 32;

  __shared__ __attribute__((aligned(16))) char smem[8 * 1024 + 32 * 1024];
  __bf16* As  = (__bf16*)smem;              // 64 x 64
  __bf16* BsT = (__bf16*)(smem + 8 * 1024); // 256n x 64k

  v8f acc[4][2];
#pragma unroll
  for (int mt = 0; mt < 4; ++mt) { acc[mt][0] = (v8f){}; acc[mt][1] = (v8f){}; }

  for (int k0 = 0; k0 < DMODEL; k0 += 64) {
#pragma unroll
    for (int j = 0; j < 2; ++j) {
      int idx = tid + j * 256;
      int r = idx >> 3, c = (idx & 7) * 8;
      copy16(&ctx[(size_t)(row0 + r) * DMODEL + k0 + c], &As[r * 64 + c]);
    }
#pragma unroll
    for (int j = 0; j < 8; ++j) {
      int idx = tid + j * 256;
      int n = idx >> 3, c = (idx & 7) * 8;
      copy16(&WoT[(size_t)(col0 + n) * DMODEL + k0 + c], &BsT[n * 64 + c]);
    }
    copy_wait();
    __syncthreads();

#pragma unroll
    for (int kk = 0; kk < 64; kk += 32) {
      Frag16 b0, b1;
      b0.u[0] = *(const uint4*)(const void*)&BsT[(n0 + m16) * 64 + kk + 16 * kh];
      b0.u[1] = *(const uint4*)(const void*)&BsT[(n0 + m16) * 64 + kk + 16 * kh + 8];
      b1.u[0] = *(const uint4*)(const void*)&BsT[(n0 + 16 + m16) * 64 + kk + 16 * kh];
      b1.u[1] = *(const uint4*)(const void*)&BsT[(n0 + 16 + m16) * 64 + kk + 16 * kh + 8];
#pragma unroll
      for (int mt = 0; mt < 4; ++mt) {
        Frag16 a;
        a.u[0] = *(const uint4*)(const void*)&As[(mt * 16 + m16) * 64 + kk + 16 * kh];
        a.u[1] = *(const uint4*)(const void*)&As[(mt * 16 + m16) * 64 + kk + 16 * kh + 8];
        acc[mt][0] = wmma_bf16(a.v, b0.v, acc[mt][0]);
        acc[mt][1] = wmma_bf16(a.v, b1.v, acc[mt][1]);
      }
    }
    __syncthreads();
  }

#pragma unroll
  for (int mt = 0; mt < 4; ++mt)
#pragma unroll
    for (int r = 0; r < 8; ++r) {
      int m = row0 + mt * 16 + r + 8 * kh;
      out[(size_t)m * DMODEL + col0 + n0 + m16]      = acc[mt][0][r];
      out[(size_t)m * DMODEL + col0 + n0 + 16 + m16] = acc[mt][1][r];
    }
}

// ---------------------------------------------------------------------------
extern "C" void kernel_launch(void* const* d_in, const int* in_sizes, int n_in,
                              void* d_out, int out_size, void* d_ws, size_t ws_size,
                              hipStream_t stream) {
  (void)in_sizes; (void)n_in; (void)out_size; (void)ws_size;
  const float* hs   = (const float*)d_in[0];
  const float* cosb = (const float*)d_in[1];
  const float* sinb = (const float*)d_in[2];
  const int*   doc  = (const int*)d_in[3];
  const float* Wq   = (const float*)d_in[4];
  const float* Wk   = (const float*)d_in[5];
  const float* Wv   = (const float*)d_in[6];
  const float* Wo   = (const float*)d_in[7];
  const float* qnw  = (const float*)d_in[8];
  const float* knw  = (const float*)d_in[9];
  float* out = (float*)d_out;

  // ws: q 16 | k 8 | vT 8 | ctx 16 | WqT 8 | WkT 4 | WvT 4 | WoT 8 | hsb 16 = 88MB
  char* ws = (char*)d_ws;
  const size_t qn  = (size_t)SEQ * NH * DHEAD;
  const size_t kvn = (size_t)SEQ * NKV * DHEAD;
  const size_t wqn = (size_t)DMODEL * NH * DHEAD;
  const size_t wkn = (size_t)DMODEL * NKV * DHEAD;
  const size_t hn  = (size_t)SEQ * DMODEL;
  __bf16* qb  = (__bf16*)ws;                 ws += qn * 2;
  __bf16* kb  = (__bf16*)ws;                 ws += kvn * 2;
  __bf16* vt  = (__bf16*)ws;                 ws += kvn * 2;
  __bf16* ctx = (__bf16*)ws;                 ws += qn * 2;
  __bf16* WqT = (__bf16*)ws;                 ws += wqn * 2;
  __bf16* WkT = (__bf16*)ws;                 ws += wkn * 2;
  __bf16* WvT = (__bf16*)ws;                 ws += wkn * 2;
  __bf16* WoT = (__bf16*)ws;                 ws += wqn * 2;
  __bf16* hsb = (__bf16*)ws;

  dim3 blk(256);
  k_cvt<<<dim3((int)(hn / 2048)), blk, 0, stream>>>(hs, hsb, (int)hn);
  k_cvt_t<<<dim3(8, 256), blk, 0, stream>>>(Wq, WqT, NH * DHEAD, DMODEL);
  k_cvt_t<<<dim3(4, 256), blk, 0, stream>>>(Wk, WkT, NKV * DHEAD, DMODEL);
  k_cvt_t<<<dim3(4, 256), blk, 0, stream>>>(Wv, WvT, NKV * DHEAD, DMODEL);
  k_cvt_t<<<dim3(8, 256), blk, 0, stream>>>(Wo, WoT, DMODEL, DMODEL);

  k_qkv<<<dim3(SEQ / 64, NH + 2 * NKV), blk, 0, stream>>>(
      hsb, cosb, sinb, WqT, WkT, WvT, qnw, knw, qb, kb, vt);
  k_attn<<<dim3(SEQ / 32, NH), blk, 0, stream>>>(qb, kb, vt, doc, ctx);
  k_oproj<<<dim3(SEQ / 64, DMODEL / 256), blk, 0, stream>>>(ctx, WoT, out);
}
